// MultiHeadAttention_26895085208030
// MI455X (gfx1250) — matrix operand order repaired and checked by reading, not compiled, not run on hardware
//
#include <hip/hip_runtime.h>
#include <hip/hip_bf16.h>

// Problem constants (from the reference)
#define D_MODEL   1024
#define NUM_HEADS 16
#define DK        64
#define SEQ       2048
#define BATCH     2
#define M_TOT     (BATCH * SEQ)   // 4096 rows for the big GEMMs

typedef __bf16 bf16_t;
typedef __attribute__((ext_vector_type(16))) __bf16 v16bf;
typedef __attribute__((ext_vector_type(8)))  __bf16 v8bf;
typedef __attribute__((ext_vector_type(4)))  __bf16 v4bf;
typedef __attribute__((ext_vector_type(8)))  float  v8f;
typedef __attribute__((ext_vector_type(4)))  float  v4f;

// ---------- helpers ----------

// float -> bf16, round-to-nearest-even (manual, avoids fptrunc-bf16 lowering)
static __device__ __forceinline__ bf16_t f2bf(float f) {
  unsigned u = __builtin_bit_cast(unsigned, f);
  unsigned r = u + 0x7FFFu + ((u >> 16) & 1u);
  unsigned short h = (unsigned short)(r >> 16);
  return __builtin_bit_cast(bf16_t, h);
}

static __device__ __forceinline__ v16bf cat8(v8bf lo, v8bf hi) {
  return __builtin_shufflevector(lo, hi, 0,1,2,3,4,5,6,7,8,9,10,11,12,13,14,15);
}

static __device__ __forceinline__ v8f wmma_bf16(v16bf a, v16bf b, v8f c) {
  // D = A(16x32 bf16) * B(32x16 bf16) + C(16x16 f32)
  return __builtin_amdgcn_wmma_f32_16x16x32_bf16(
      /*neg_a=*/false, a, /*neg_b=*/false, b,
      /*c_mod=*/(short)0, c, /*reuse_a=*/false, /*reuse_b=*/false);
}

// A-fragment (16x32 bf16): lane L holds row (L&15),
// K = {2*kb..2*kb+15}, kb = 8*(L>>4) (contiguous, matches B).
static __device__ __forceinline__ v16bf load_a_frag(const bf16_t* rowptr, int kbase) {
  v8bf lo = *(const v8bf*)(rowptr + 2 * kbase);
  v8bf hi = *(const v8bf*)(rowptr + 2 * kbase + 8);
  return cat8(lo, hi);
}

// 16-byte async global->LDS copy (CDNA5 GLOBAL_LOAD_ASYNC_TO_LDS_B128,
// tracked by ASYNCcnt).  vdst = LDS byte address (generic LDS addr[31:0]),
// vaddr = 64-bit global address, saddr = off.
static __device__ __forceinline__ void cp_async16(bf16_t* lds_dst, const bf16_t* gsrc) {
  unsigned lds_off = (unsigned)(uintptr_t)lds_dst;   // LDS aperture: addr[31:0]
  asm volatile("global_load_async_to_lds_b128 %0, %1, off"
               :: "v"(lds_off), "v"(gsrc) : "memory");
}

// ---------- kernels ----------

// fp32 -> bf16, 4 elements per thread
__global__ void cvt_f32_to_bf16(const float* __restrict__ src,
                                bf16_t* __restrict__ dst, int n4) {
  int i = blockIdx.x * blockDim.x + threadIdx.x;
  if (i >= n4) return;
  v4f f = *(const v4f*)(src + 4 * (size_t)i);
  v4bf o;
  o[0] = f2bf(f[0]); o[1] = f2bf(f[1]); o[2] = f2bf(f[2]); o[3] = f2bf(f[3]);
  *(v4bf*)(dst + 4 * (size_t)i) = o;
}

// C[M,N] = (A[M,K] * W[N,K]^T + bias) * scale  (nn.Linear convention),
// output bf16 head-split:
//   transposed == 0 : out[((b*H + h)*SEQ + s)*DK + d]   (q, k)
//   transposed == 1 : out[((b*H + h)*DK  + d)*SEQ + s]  (v, pre-transposed)
// WG = 256 threads = 8 waves; wave tile 32x64 (2 A-frags x 4 B-frags,
// 8 WMMAs per k-step); WG tile 64x256.
__global__ __launch_bounds__(256) void gemm_proj(
    const bf16_t* __restrict__ A, const bf16_t* __restrict__ W,
    const float* __restrict__ bias, bf16_t* __restrict__ out,
    float scale, int transposed) {
  const int K = D_MODEL;
  const int lane = threadIdx.x & 31;
  const int wid  = threadIdx.x >> 5;
  const int half = lane >> 4, cl = lane & 15;
  const int row0 = blockIdx.y * 64  + (wid & 1) * 32;
  const int col0 = blockIdx.x * 256 + (wid >> 1) * 64;

  v8f z = {0.f,0.f,0.f,0.f,0.f,0.f,0.f,0.f};
  v8f acc0[4] = {z, z, z, z};
  v8f acc1[4] = {z, z, z, z};

  const bf16_t* arow0 = A + (size_t)(row0 +      cl) * K;
  const bf16_t* arow1 = A + (size_t)(row0 + 16 + cl) * K;
  const int kb = 8 * half;
  for (int k0 = 0; k0 < K; k0 += 32) {
    __builtin_prefetch(arow0 + k0 + 256, 0, 3);   // global_prefetch_b8
    v16bf a0 = load_a_frag(arow0 + k0, kb);
    v16bf a1 = load_a_frag(arow1 + k0, kb);
    v16bf bfr[4];
#pragma unroll
    for (int nt = 0; nt < 4; ++nt)                 // clause of 4 x 32B loads
      bfr[nt] = *(const v16bf*)(W + (size_t)(col0 + nt * 16 + cl) * K + k0 + 16 * half);
#pragma unroll
    for (int nt = 0; nt < 4; ++nt) {               // 8 WMMAs, B reused twice
      acc0[nt] = wmma_bf16(a0, bfr[nt], acc0[nt]);
      acc1[nt] = wmma_bf16(a1, bfr[nt], acc1[nt]);
    }
  }

#pragma unroll
  for (int nt = 0; nt < 4; ++nt) {
    const int col = col0 + nt * 16 + cl;
    const float bv = bias[col];
    const int h = col >> 6, d = col & 63;
#pragma unroll
    for (int r = 0; r < 8; ++r) {
#pragma unroll
      for (int half2 = 0; half2 < 2; ++half2) {
        const int row = row0 + 16 * half2 + r + 8 * half;
        const int b = row / SEQ, s = row % SEQ;
        const float v = ((half2 ? acc1[nt][r] : acc0[nt][r]) + bv) * scale;
        const size_t idx = transposed
            ? ((size_t)((b * NUM_HEADS + h) * DK + d)) * SEQ + s
            : ((size_t)((b * NUM_HEADS + h) * SEQ + s)) * DK + d;
        out[idx] = f2bf(v);
      }
    }
  }
}

// out[M,N] fp32 = A[M,K]bf16 * W[N,K]^T + bias  (final projection)
__global__ __launch_bounds__(256) void gemm_out(
    const bf16_t* __restrict__ A, const bf16_t* __restrict__ W,
    const float* __restrict__ bias, float* __restrict__ out) {
  const int K = D_MODEL;
  const int lane = threadIdx.x & 31;
  const int wid  = threadIdx.x >> 5;
  const int half = lane >> 4, cl = lane & 15;
  const int row0 = blockIdx.y * 64  + (wid & 1) * 32;
  const int col0 = blockIdx.x * 256 + (wid >> 1) * 64;

  v8f z = {0.f,0.f,0.f,0.f,0.f,0.f,0.f,0.f};
  v8f acc0[4] = {z, z, z, z};
  v8f acc1[4] = {z, z, z, z};

  const bf16_t* arow0 = A + (size_t)(row0 +      cl) * K;
  const bf16_t* arow1 = A + (size_t)(row0 + 16 + cl) * K;
  const int kb = 8 * half;
  for (int k0 = 0; k0 < K; k0 += 32) {
    __builtin_prefetch(arow0 + k0 + 256, 0, 3);
    v16bf a0 = load_a_frag(arow0 + k0, kb);
    v16bf a1 = load_a_frag(arow1 + k0, kb);
    v16bf bfr[4];
#pragma unroll
    for (int nt = 0; nt < 4; ++nt)
      bfr[nt] = *(const v16bf*)(W + (size_t)(col0 + nt * 16 + cl) * K + k0 + 16 * half);
#pragma unroll
    for (int nt = 0; nt < 4; ++nt) {
      acc0[nt] = wmma_bf16(a0, bfr[nt], acc0[nt]);
      acc1[nt] = wmma_bf16(a1, bfr[nt], acc1[nt]);
    }
  }

#pragma unroll
  for (int nt = 0; nt < 4; ++nt) {
    const int col = col0 + nt * 16 + cl;
    const float bv = bias[col];
#pragma unroll
    for (int r = 0; r < 8; ++r) {
      const int rowA = row0 + r + 8 * half;
      out[(size_t)rowA * D_MODEL + col] = acc0[nt][r] + bv;
      out[(size_t)(rowA + 16) * D_MODEL + col] = acc1[nt][r] + bv;
    }
  }
}

// Flash attention per (b,h).  WG = 256 threads = 8 waves; each wave owns
// 16 query rows x full DK=64 (128 q rows per WG).  K/V tiles (32 kv
// positions) are staged in LDS with async global->LDS copies, double-
// buffered; all 8 waves share them (8x reuse of global K/V traffic).
// q,k in [B,H,S,DK]; v pre-transposed [B,H,DK,S]; out attn bf16 [B,S,D].
__global__ __launch_bounds__(256) void flash_attn(
    const bf16_t* __restrict__ qh, const bf16_t* __restrict__ kh,
    const bf16_t* __restrict__ vt, bf16_t* __restrict__ attn) {
  __shared__ bf16_t Kt[2][32][DK];     // 8 KB: K tile,  [kv][dk]
  __shared__ bf16_t Vs[2][DK][32];     // 8 KB: V^T tile, [d][kv]
  __shared__ bf16_t Pb[8][16][32];     // 8 KB: per-wave P bounce

  const int tid = threadIdx.x;
  const int bh = blockIdx.y;                      // b*H + h
  const int b = bh >> 4, h = bh & 15;
  const int lane = tid & 31, wid = tid >> 5;
  const int half = lane >> 4, cl = lane & 15;
  const int qrow0 = blockIdx.x * 128 + wid * 16;

  const bf16_t* Qh = qh + (size_t)bh * SEQ * DK;
  const bf16_t* Kh = kh + (size_t)bh * SEQ * DK;
  const bf16_t* Vg = vt + (size_t)bh * DK * SEQ;
  bf16_t* pw = &Pb[wid][0][0];

  // issue async copies for one 32-kv stage: 4 KB K + 4 KB V = 512 x 16B
  // chunks over 256 threads -> 2 async b128 per thread (1 K + 1 V)
  auto stage = [&](int buf, int kv0) {
    const int kr = tid >> 3, kc8 = (tid & 7) * 8;   // K: 32 rows x 8 chunks
    cp_async16(&Kt[buf][kr][kc8], Kh + (size_t)(kv0 + kr) * DK + kc8);
    const int vr = tid >> 2, vc8 = (tid & 3) * 8;   // V: 64 rows x 4 chunks
    cp_async16(&Vs[buf][vr][vc8], Vg + (size_t)vr * SEQ + kv0 + vc8);
  };

  // Q fragments for this wave's 16 rows (1/sqrt(dk) folded into q projection)
  const bf16_t* qrow = Qh + (size_t)(qrow0 + cl) * DK;
  const int kb = 8 * half;
  v16bf qa0 = load_a_frag(qrow +  0, kb);
  v16bf qa1 = load_a_frag(qrow + 32, kb);

  v8f z = {0.f,0.f,0.f,0.f,0.f,0.f,0.f,0.f};
  v8f o[4] = {z, z, z, z};
  float mstat[8], lstat[8];
#pragma unroll
  for (int r = 0; r < 8; ++r) { mstat[r] = -1e30f; lstat[r] = 0.f; }

  stage(0, 0);
  asm volatile("s_wait_asynccnt 0" ::: "memory");
  __syncthreads();

  for (int kv0 = 0; kv0 < SEQ; kv0 += 32) {
    const int buf = (kv0 >> 5) & 1;
    const bool more = (kv0 + 32) < SEQ;            // uniform
    if (more) stage(buf ^ 1, kv0 + 32);            // overlap copy with compute

    // ---- scores: two 16x16 tiles over kv, contracted over DK (from LDS) ----
    v8f s0 = z, s1 = z;
    {
      v16bf k00 = *(const v16bf*)&Kt[buf][     cl][ 0 + 16 * half];
      v16bf k10 = *(const v16bf*)&Kt[buf][16 + cl][ 0 + 16 * half];
      s0 = wmma_bf16(qa0, k00, s0);
      s1 = wmma_bf16(qa0, k10, s1);
      v16bf k01 = *(const v16bf*)&Kt[buf][     cl][32 + 16 * half];
      v16bf k11 = *(const v16bf*)&Kt[buf][16 + cl][32 + 16 * half];
      s0 = wmma_bf16(qa1, k01, s0);
      s1 = wmma_bf16(qa1, k11, s1);
    }

    // ---- online softmax; row r+8*half lives across the 16 lanes of a half ----
#pragma unroll
    for (int r = 0; r < 8; ++r) {
      float mx = fmaxf(s0[r], s1[r]);
      mx = fmaxf(mx, __shfl_xor(mx, 1, 32));
      mx = fmaxf(mx, __shfl_xor(mx, 2, 32));
      mx = fmaxf(mx, __shfl_xor(mx, 4, 32));
      mx = fmaxf(mx, __shfl_xor(mx, 8, 32));
      const float mnew  = fmaxf(mstat[r], mx);
      const float alpha = __expf(mstat[r] - mnew);
      mstat[r] = mnew;
      const float p0 = __expf(s0[r] - mnew);
      const float p1 = __expf(s1[r] - mnew);
      float sum = p0 + p1;
      sum += __shfl_xor(sum, 1, 32);
      sum += __shfl_xor(sum, 2, 32);
      sum += __shfl_xor(sum, 4, 32);
      sum += __shfl_xor(sum, 8, 32);
      lstat[r] = lstat[r] * alpha + sum;
      o[0][r] *= alpha; o[1][r] *= alpha; o[2][r] *= alpha; o[3][r] *= alpha;
      s0[r] = p0; s1[r] = p1;
    }

    // ---- P: C-layout -> A-fragment layout via per-wave LDS bounce ----
#pragma unroll
    for (int r = 0; r < 8; ++r) {
      const int prow = r + 8 * half;
      pw[prow * 32 +      cl] = f2bf(s0[r]);
      pw[prow * 32 + 16 + cl] = f2bf(s1[r]);
    }
    asm volatile("s_wait_dscnt 0" ::: "memory");   // intra-wave LDS RAW fence
    v16bf pa = cat8(*(const v8bf*)&pw[cl * 32 + 2 * kb],
                    *(const v8bf*)&pw[cl * 32 + 2 * kb + 8]);

    // ---- O += P * V  (kv chunk of 32; V^T rows contiguous in LDS) ----
#pragma unroll
    for (int nt = 0; nt < 4; ++nt) {
      v16bf vb = *(const v16bf*)&Vs[buf][nt * 16 + cl][16 * half];
      o[nt] = wmma_bf16(pa, vb, o[nt]);
    }

    if (more) {
      asm volatile("s_wait_asynccnt 0" ::: "memory");  // next tile landed
      __syncthreads();                                  // publish across waves
    }
  }

  // ---- epilogue: normalize, store attn [B,S,D] bf16 ----
#pragma unroll
  for (int r = 0; r < 8; ++r) {
    const float inv = lstat[r] > 0.f ? 1.f / lstat[r] : 0.f;
    const int s = qrow0 + r + 8 * half;
    const size_t base = ((size_t)(b * SEQ + s)) * D_MODEL + h * DK;
#pragma unroll
    for (int nt = 0; nt < 4; ++nt)
      attn[base + nt * 16 + cl] = f2bf(o[nt][r] * inv);
  }
}

// ---------- host ----------

extern "C" void kernel_launch(void* const* d_in, const int* in_sizes, int n_in,
                              void* d_out, int out_size, void* d_ws, size_t ws_size,
                              hipStream_t stream) {
  const float* Q  = (const float*)d_in[0];
  const float* K  = (const float*)d_in[1];
  const float* V  = (const float*)d_in[2];
  const float* Wq = (const float*)d_in[3];
  const float* bq = (const float*)d_in[4];
  const float* Wk = (const float*)d_in[5];
  const float* bk = (const float*)d_in[6];
  const float* Wv = (const float*)d_in[7];
  const float* bv = (const float*)d_in[8];
  const float* Wo = (const float*)d_in[9];
  const float* bo = (const float*)d_in[10];
  float* out = (float*)d_out;

  char* ws = (char*)d_ws;
  size_t off = 0;
  auto alloc = [&](size_t bytes) { char* p = ws + off; off += bytes; return p; };

  const size_t act_b = (size_t)M_TOT * D_MODEL * sizeof(bf16_t);   // 8 MiB
  const size_t wgt_b = (size_t)D_MODEL * D_MODEL * sizeof(bf16_t); // 2 MiB

  bf16_t* Qb   = (bf16_t*)alloc(act_b);
  bf16_t* Kb   = (bf16_t*)alloc(act_b);
  bf16_t* Vb   = (bf16_t*)alloc(act_b);
  bf16_t* Wqb  = (bf16_t*)alloc(wgt_b);
  bf16_t* Wkb  = (bf16_t*)alloc(wgt_b);
  bf16_t* Wvb  = (bf16_t*)alloc(wgt_b);
  bf16_t* Wob  = (bf16_t*)alloc(wgt_b);
  bf16_t* qhh  = (bf16_t*)alloc(act_b);   // [B,H,S,DK]
  bf16_t* khh  = (bf16_t*)alloc(act_b);   // [B,H,S,DK]
  bf16_t* vtt  = (bf16_t*)alloc(act_b);   // [B,H,DK,S]
  bf16_t* attn = (bf16_t*)alloc(act_b);   // [B,S,D]

  const int na4 = M_TOT * D_MODEL / 4;
  const int nw4 = D_MODEL * D_MODEL / 4;
  cvt_f32_to_bf16<<<dim3((na4 + 255) / 256), 256, 0, stream>>>(Q, Qb, na4);
  cvt_f32_to_bf16<<<dim3((na4 + 255) / 256), 256, 0, stream>>>(K, Kb, na4);
  cvt_f32_to_bf16<<<dim3((na4 + 255) / 256), 256, 0, stream>>>(V, Vb, na4);
  cvt_f32_to_bf16<<<dim3((nw4 + 255) / 256), 256, 0, stream>>>(Wq, Wqb, nw4);
  cvt_f32_to_bf16<<<dim3((nw4 + 255) / 256), 256, 0, stream>>>(Wk, Wkb, nw4);
  cvt_f32_to_bf16<<<dim3((nw4 + 255) / 256), 256, 0, stream>>>(Wv, Wvb, nw4);
  cvt_f32_to_bf16<<<dim3((nw4 + 255) / 256), 256, 0, stream>>>(Wo, Wob, nw4);

  const dim3 ggemm(D_MODEL / 256, M_TOT / 64);   // (4, 64)
  const float scale = 0.125f;                     // 1/sqrt(DK), folded into q
  gemm_proj<<<ggemm, 256, 0, stream>>>(Qb, Wqb, bq, qhh, scale, 0);
  gemm_proj<<<ggemm, 256, 0, stream>>>(Kb, Wkb, bk, khh, 1.0f, 0);
  gemm_proj<<<ggemm, 256, 0, stream>>>(Vb, Wvb, bv, vtt, 1.0f, 1);

  const dim3 gfa(SEQ / 128, BATCH * NUM_HEADS);   // (16, 32)
  flash_attn<<<gfa, 256, 0, stream>>>(qhh, khh, vtt, attn);

  gemm_out<<<ggemm, 256, 0, stream>>>(attn, Wob, bo, out);
}
